// MultiModalFakeNewsGNN_79826262163921
// MI455X (gfx1250) — matrix operand order repaired and checked by reading, not compiled, not run on hardware
//
#include <hip/hip_runtime.h>
#include <hip/hip_bf16.h>
#include <stdint.h>

// ---------------- problem constants (from reference) ----------------
#define N_NODES  50000
#define E_EDGES  1600000
#define EP_EDGES (E_EDGES + N_NODES)   // edges + self loops
#define IN_DIM   1280
#define HID      256
#define HEADS    4
#define DH       64
#define NCLS     6
#define C1DIM    128
#define NEG_SLOPE 0.2f
#define LN_EPS    1e-5f

typedef __attribute__((ext_vector_type(16))) _Float16 v16h;
typedef __attribute__((ext_vector_type(8)))  float    v8f;

// =====================================================================
// Fragment packing (CDNA5 ISA layouts).
// A 16x32 f16: lanes 0-15 M=lane, K pairs 0..7|16..23; lanes 16-31 +8 on K.
//   Packed: [mTile][kChunk][lane(32)][half(16)] -> 32B contiguous per lane.
// B 32x16 f16: lane = column (dup in 16-31, +8 on K), half j -> K 0..7|16..23.
//   Packed: [kChunk][nTile][lane(32)][half(16)].
// =====================================================================
__global__ void pack_a_f16(const float* __restrict__ src, _Float16* __restrict__ dst,
                           int M, int K)
{
    size_t tid = (size_t)blockIdx.x * blockDim.x + threadIdx.x;
    if (tid >= (size_t)M * K) return;
    int j    = (int)(tid & 15);
    int lane = (int)((tid >> 4) & 31);
    size_t tl = tid >> 9;                 // tile-linear = mTile*kChunks + kc
    int kChunks = K >> 5;
    int kc    = (int)(tl % kChunks);
    int mTile = (int)(tl / kChunks);
    int dm = lane & 15;
    int hi = lane >> 4;
    int v = j >> 1, p = j & 1;
    int k = ((v < 4) ? (v * 2 + p) : (16 + (v - 4) * 2 + p)) + hi * 8;
    dst[tid] = (_Float16)src[(size_t)(mTile * 16 + dm) * K + kc * 32 + k];
}

__global__ void pack_b_f16(const float* __restrict__ src, _Float16* __restrict__ dst,
                           int K, int Ncols)
{
    size_t tid = (size_t)blockIdx.x * blockDim.x + threadIdx.x;
    if (tid >= (size_t)K * Ncols) return;
    int j    = (int)(tid & 15);
    int lane = (int)((tid >> 4) & 31);
    size_t tl = tid >> 9;                 // tile-linear = kc*nTiles + nTile
    int nTiles = Ncols >> 4;
    int nTile = (int)(tl % nTiles);
    int kc    = (int)(tl / nTiles);
    int n = lane & 15;
    int k = ((j < 8) ? j : (16 + (j - 8))) + ((lane >> 4) << 3);
    dst[tid] = (_Float16)src[(size_t)(kc * 32 + k) * Ncols + nTile * 16 + n];
}

// =====================================================================
// B staging: one 4 KB k-chunk (4 n-tile fragments) into LDS.
// Preferred path: Tensor Data Mover (single descriptor per block per
// k-chunk, tracked by TENSORcnt).  Fallback: per-thread CDNA5 async
// global->LDS copies (ASYNCcnt).
// =====================================================================
#if defined(__AMDGCN__) && __has_builtin(__builtin_amdgcn_tensor_load_to_lds)
#define USE_TDM 1
typedef __attribute__((ext_vector_type(4))) unsigned int u32x4;
typedef __attribute__((ext_vector_type(8))) int          i32x8;
typedef __attribute__((ext_vector_type(4))) int          i32x4;

// D# per CDNA5 ISA: 1-row 2D tile, data_size=8B,
// tile_dim0 = 512 elements = 4096 bytes.
__device__ __forceinline__ void tdm_stage_b(const _Float16* __restrict__ Bp,
                                            _Float16* __restrict__ bshBuf,
                                            int kc, int nTiles, int nG)
{
    unsigned long long ga = (unsigned long long)(uintptr_t)
        (Bp + (((size_t)kc * nTiles + (nG << 2)) * 512));
    unsigned int lds = (unsigned int)(uintptr_t)bshBuf;   // flat[31:0] == LDS offset
    u32x4 g0;
    g0.x = 1u;                                            // count=1 (valid user D#)
    g0.y = lds;                                           // lds_addr
    g0.z = (unsigned int)(ga & 0xFFFFFFFFu);              // global_addr[31:0]
    g0.w = (unsigned int)((ga >> 32) & 0x1FFFFFFu)        // global_addr[56:32]
         | (2u << 30);                                    // type=2 ("image")
    i32x8 g1;
    g1[0] = (3 << 16);       // workgroup_mask=0 | data_size=3 (8B) | no flags
    g1[1] = (512 << 16);     // atomic_barrier_addr=0 | tensor_dim0[15:0]=512
    g1[2] = (1 << 16);       // tensor_dim0[31:16]=0 | tensor_dim1[15:0]=1
    g1[3] = (512 << 16);     // tensor_dim1[31:16]=0 | tile_dim0=512
    g1[4] = 0;               // tile_dim1=0 (unused) | tile_dim2=0
    g1[5] = 512;             // tensor_dim0_stride[31:0]=512
    g1[6] = 0;               // stride0[47:32]=0 | stride1[15:0]=0
    g1[7] = 0;
    i32x4 z4 = {0, 0, 0, 0};                 // groups 2/3 unused (<=2D tensor)
    i32x8 z8 = {0, 0, 0, 0, 0, 0, 0, 0};     // trailing group (6-arg toolchain)
    __builtin_amdgcn_tensor_load_to_lds(g0, g1, z4, z4, z8, 0);
}
#endif

__device__ __forceinline__ void stage_b_async(const _Float16* __restrict__ Bp,
                                              _Float16* __restrict__ bshBuf,
                                              int kc, int nTiles, int nG, int tid)
{
    const _Float16* src = Bp + (((size_t)kc * nTiles + (nG << 2)) * 512) + (size_t)tid * 8;
    uint32_t lds_addr = (uint32_t)(uintptr_t)(bshBuf + (size_t)tid * 8);
    asm volatile("global_load_async_to_lds_b128 %0, %1, off"
                 :: "v"(lds_addr), "v"((unsigned long long)(uintptr_t)src)
                 : "memory");
}

__device__ __forceinline__ void wait_asynccnt0()
{
    asm volatile("s_wait_asynccnt 0" ::: "memory");
}

// =====================================================================
// WMMA GEMM, LDS-staged B with double buffering.
// Block = 256 threads = 8 waves; wave w computes m-tile (mBlock*8+w) for
// one group of 4 n-tiles; B fragments staged to LDS (8 KB) by TDM (or
// async copies), A streamed from global with prefetch.
// =====================================================================
__device__ __forceinline__ void store_tile(float* __restrict__ C, int Ncols,
                                           int mBase, int col, v8f acc,
                                           const float* __restrict__ bias, int relu)
{
    float bv = bias ? bias[col] : 0.0f;
#pragma unroll
    for (int r = 0; r < 8; ++r) {
        float v = acc[r] + bv;
        if (relu) v = fmaxf(v, 0.0f);
        C[(size_t)(mBase + r) * Ncols + col] = v;
    }
}

__global__ void gemm_wmma_f16(const _Float16* __restrict__ Ap,
                              const _Float16* __restrict__ Bp,
                              const float* __restrict__ bias,
                              float* __restrict__ C,
                              int mTiles, int nTiles, int kChunks, int Ncols, int relu)
{
    __shared__ _Float16 Bsh[2][4 * 512];          // 2 x 4KB double buffer

    const int tid  = threadIdx.x;
    const int lane = tid & 31;
    const int w    = tid >> 5;
    const int nGroups = nTiles >> 2;
    const int mBlock  = blockIdx.x / nGroups;
    const int nG      = blockIdx.x % nGroups;
    const int mTile   = mBlock * 8 + w;
    const bool active = (mTile < mTiles);          // wave-uniform

    v8f a0 = {}, a1 = {}, a2 = {}, a3 = {};
    const _Float16* aPtr = Ap + ((size_t)mTile * kChunks) * 512 + (size_t)lane * 16;

#ifdef USE_TDM
    if (w == 0) tdm_stage_b(Bp, &Bsh[0][0], 0, nTiles, nG);
#else
    stage_b_async(Bp, &Bsh[0][0], 0, nTiles, nG, tid);
#endif

    int buf = 0;
    for (int kc = 0; kc < kChunks; ++kc) {
#ifdef USE_TDM
        if (w == 0) __builtin_amdgcn_s_wait_tensorcnt(0); // DMA into Bsh[buf] done
#else
        wait_asynccnt0();
#endif
        __syncthreads();           // writes visible; prior reads of buf^1 retired
        if (kc + 1 < kChunks) {
#ifdef USE_TDM
            if (w == 0) tdm_stage_b(Bp, &Bsh[buf ^ 1][0], kc + 1, nTiles, nG);
#else
            stage_b_async(Bp, &Bsh[buf ^ 1][0], kc + 1, nTiles, nG, tid);
#endif
        }

        if (active) {
            v16h a = *(const v16h*)aPtr;
            if (kc + 1 < kChunks) __builtin_prefetch(aPtr + 512, 0, 3); // global_prefetch_b8
            const _Float16* bb = &Bsh[buf][0] + (size_t)lane * 16;
            v16h b0 = *(const v16h*)(bb);                // ds_load_b128 x2
            v16h b1 = *(const v16h*)(bb + 512);
            v16h b2 = *(const v16h*)(bb + 1024);
            v16h b3 = *(const v16h*)(bb + 1536);
            a0 = __builtin_amdgcn_wmma_f32_16x16x32_f16(false, a, false, b0, (short)0, a0, false, false);
            a1 = __builtin_amdgcn_wmma_f32_16x16x32_f16(false, a, false, b1, (short)0, a1, false, false);
            a2 = __builtin_amdgcn_wmma_f32_16x16x32_f16(false, a, false, b2, (short)0, a2, false, false);
            a3 = __builtin_amdgcn_wmma_f32_16x16x32_f16(false, a, false, b3, (short)0, a3, false, false);
        }
        aPtr += 512;
        buf ^= 1;
    }

    if (active) {
        const int mBase = mTile * 16 + ((lane >> 4) << 3);
        const int col   = (nG << 2) * 16 + (lane & 15);
        store_tile(C, Ncols, mBase, col,      a0, bias, relu);
        store_tile(C, Ncols, mBase, col + 16, a1, bias, relu);
        store_tile(C, Ncols, mBase, col + 32, a2, bias, relu);
        store_tile(C, Ncols, mBase, col + 48, a3, bias, relu);
    }
}

// =====================================================================
// Per-node attention dots: warp per (node, head); 2 elems/lane, shfl reduce.
// =====================================================================
__global__ void att_dots(const float* __restrict__ xh,
                         const float* __restrict__ asrc,
                         const float* __restrict__ adst,
                         float* __restrict__ alsrc,
                         float* __restrict__ aldst)
{
    int g = blockIdx.x * blockDim.x + threadIdx.x;
    int w = g >> 5, lane = g & 31;
    int node = w >> 2, head = w & 3;
    if (node >= N_NODES) return;
    int k = lane * 2;
    const float* xr = xh + (size_t)node * HID + head * DH;
    float x0 = xr[k], x1 = xr[k + 1];
    float s = x0 * asrc[head * DH + k] + x1 * asrc[head * DH + k + 1];
    float t = x0 * adst[head * DH + k] + x1 * adst[head * DH + k + 1];
#pragma unroll
    for (int m = 16; m >= 1; m >>= 1) {
        s += __shfl_xor(s, m, 32);
        t += __shfl_xor(t, m, 32);
    }
    if (lane == 0) {
        alsrc[node * 4 + head] = s;
        aldst[node * 4 + head] = t;
    }
}

__global__ void fill_u32(uint32_t* __restrict__ p, uint32_t v, size_t n)
{
    size_t t = (size_t)blockIdx.x * blockDim.x + threadIdx.x;
    if (t < n) p[t] = v;
}

__device__ __forceinline__ void atomicMaxFloat(float* addr, float v)
{
    // signed/unsigned split trick; init must be 0xFF800000 (-inf)
    if (!(v < 0.0f)) atomicMax((int*)addr, __float_as_int(v));
    else             atomicMin((unsigned int*)addr, __float_as_uint(v));
}

__device__ __forceinline__ void edge_ends(int e, const long long* __restrict__ ei,
                                          int& s, int& d)
{
    if (e < E_EDGES) { s = (int)ei[e]; d = (int)ei[(size_t)E_EDGES + e]; }
    else             { s = d = e - E_EDGES; }   // self loop
}

__global__ void edge_softmax_max(const long long* __restrict__ ei,
                                 const float* __restrict__ alsrc,
                                 const float* __restrict__ aldst,
                                 float* __restrict__ mmax)
{
    int e = blockIdx.x * blockDim.x + threadIdx.x;
    if (e >= EP_EDGES) return;
    int s, d; edge_ends(e, ei, s, d);
#pragma unroll
    for (int h = 0; h < 4; ++h) {
        float v = alsrc[s * 4 + h] + aldst[d * 4 + h];
        v = v > 0.0f ? v : NEG_SLOPE * v;            // leaky_relu
        atomicMaxFloat(&mmax[d * 4 + h], v);
    }
}

__global__ void edge_softmax_exp(const long long* __restrict__ ei,
                                 const float* __restrict__ alsrc,
                                 const float* __restrict__ aldst,
                                 const float* __restrict__ mmax,
                                 float* __restrict__ e_exp,
                                 float* __restrict__ denom)
{
    int e = blockIdx.x * blockDim.x + threadIdx.x;
    if (e >= EP_EDGES) return;
    int s, d; edge_ends(e, ei, s, d);
#pragma unroll
    for (int h = 0; h < 4; ++h) {
        float v = alsrc[s * 4 + h] + aldst[d * 4 + h];
        v = v > 0.0f ? v : NEG_SLOPE * v;
        float ex = __expf(v - mmax[d * 4 + h]);
        e_exp[(size_t)e * 4 + h] = ex;
        atomicAdd(&denom[d * 4 + h], ex);
    }
}

// wave per edge: 8 features/lane, f32 atomic scatter (L2-resident accumulator)
__global__ void edge_scatter(const long long* __restrict__ ei,
                             const float* __restrict__ xh,
                             const float* __restrict__ e_exp,
                             const float* __restrict__ denom,
                             float* __restrict__ outac)
{
    int g = blockIdx.x * blockDim.x + threadIdx.x;
    int e = g >> 5, lane = g & 31;
    if (e >= EP_EDGES) return;
    int s, d; edge_ends(e, ei, s, d);
    float4 ex = ((const float4*)e_exp)[e];
    float4 dn = ((const float4*)denom)[d];
    float w[4] = { ex.x / dn.x, ex.y / dn.y, ex.z / dn.z, ex.w / dn.w };
    float wh = w[lane >> 3];                 // 8 lanes per head (DH=64)
    const float4* xr = (const float4*)(xh + (size_t)s * HID + lane * 8);
    float4 x0 = xr[0], x1 = xr[1];
    float* orow = outac + (size_t)d * HID + lane * 8;
    atomicAdd(&orow[0], x0.x * wh);
    atomicAdd(&orow[1], x0.y * wh);
    atomicAdd(&orow[2], x0.z * wh);
    atomicAdd(&orow[3], x0.w * wh);
    atomicAdd(&orow[4], x1.x * wh);
    atomicAdd(&orow[5], x1.y * wh);
    atomicAdd(&orow[6], x1.z * wh);
    atomicAdd(&orow[7], x1.w * wh);
}

// h += relu(LN(outac + b_gat)) ; wave per node, 8 strided elems/lane
__global__ void ln_relu_residual(const float* __restrict__ outac,
                                 const float* __restrict__ bias,
                                 const float* __restrict__ gamma,
                                 const float* __restrict__ beta,
                                 float* __restrict__ h)
{
    int g = blockIdx.x * blockDim.x + threadIdx.x;
    int node = g >> 5, lane = g & 31;
    if (node >= N_NODES) return;
    const float* row = outac + (size_t)node * HID;
    float v[8], sum = 0.0f, sq = 0.0f;
#pragma unroll
    for (int j = 0; j < 8; ++j) {
        int f = j * 32 + lane;
        v[j] = row[f] + bias[f];
        sum += v[j]; sq += v[j] * v[j];
    }
#pragma unroll
    for (int m = 16; m >= 1; m >>= 1) {
        sum += __shfl_xor(sum, m, 32);
        sq  += __shfl_xor(sq,  m, 32);
    }
    float mu  = sum * (1.0f / HID);
    float var = sq * (1.0f / HID) - mu * mu;
    float r   = rsqrtf(var + LN_EPS);
    float* hrow = h + (size_t)node * HID;
#pragma unroll
    for (int j = 0; j < 8; ++j) {
        int f = j * 32 + lane;
        float t = (v[j] - mu) * r * gamma[f] + beta[f];
        hrow[f] += fmaxf(t, 0.0f);
    }
}

// final tiny GEMM: [N,128] x [128,6] + b, thread per node (W_c2 cache-resident)
__global__ void classifier2(const float* __restrict__ c1,
                            const float* __restrict__ W2,
                            const float* __restrict__ b2,
                            float* __restrict__ out)
{
    int node = blockIdx.x * blockDim.x + threadIdx.x;
    if (node >= N_NODES) return;
    float acc[NCLS];
#pragma unroll
    for (int j = 0; j < NCLS; ++j) acc[j] = b2[j];
    const float* row = c1 + (size_t)node * C1DIM;
    for (int k = 0; k < C1DIM; ++k) {
        float c = row[k];
#pragma unroll
        for (int j = 0; j < NCLS; ++j) acc[j] += c * W2[k * NCLS + j];
    }
#pragma unroll
    for (int j = 0; j < NCLS; ++j) out[(size_t)node * NCLS + j] = acc[j];
}

// =====================================================================
extern "C" void kernel_launch(void* const* d_in, const int* in_sizes, int n_in,
                              void* d_out, int out_size, void* d_ws, size_t ws_size,
                              hipStream_t stream)
{
    const float*     x      = (const float*)d_in[0];
    const long long* ei     = (const long long*)d_in[1];   // int64 per reference
    const float*     W_in   = (const float*)d_in[2];
    const float*     b_in   = (const float*)d_in[3];
    const float* W_gat[2]   = {(const float*)d_in[4],  (const float*)d_in[10]};
    const float* attS[2]    = {(const float*)d_in[5],  (const float*)d_in[11]};
    const float* attD[2]    = {(const float*)d_in[6],  (const float*)d_in[12]};
    const float* b_gat[2]   = {(const float*)d_in[7],  (const float*)d_in[13]};
    const float* ln_g[2]    = {(const float*)d_in[8],  (const float*)d_in[14]};
    const float* ln_b[2]    = {(const float*)d_in[9],  (const float*)d_in[15]};
    const float*     W_c1   = (const float*)d_in[16];
    const float*     b_c1   = (const float*)d_in[17];
    const float*     W_c2   = (const float*)d_in[18];
    const float*     b_c2   = (const float*)d_in[19];
    float*           logits = (float*)d_out;

    // ---------------- workspace arena ----------------
    char*  base = (char*)d_ws;
    size_t off  = 0;
    auto carve = [&](size_t bytes) -> char* {
        char* p = base + off;
        off += (bytes + 255) & ~(size_t)255;
        return p;
    };
    _Float16* XP    = (_Float16*)carve((size_t)N_NODES * IN_DIM * 2); // 128 MB, dead after GEMM1
    float*    h     = (float*)carve((size_t)N_NODES * HID * 4);
    float*    xh    = (float*)carve((size_t)N_NODES * HID * 4);
    float*    outac = (float*)carve((size_t)N_NODES * HID * 4);
    _Float16* HP    = (_Float16*)carve((size_t)N_NODES * HID * 2);
    _Float16* WinP  = (_Float16*)carve((size_t)IN_DIM * HID * 2);
    _Float16* WgP0  = (_Float16*)carve((size_t)HID * HID * 2);
    _Float16* WgP1  = (_Float16*)carve((size_t)HID * HID * 2);
    _Float16* Wc1P  = (_Float16*)carve((size_t)HID * C1DIM * 2);
    float*    alsrc = (float*)carve((size_t)N_NODES * 4 * 4);
    float*    aldst = (float*)carve((size_t)N_NODES * 4 * 4);
    float*    mmax  = (float*)carve((size_t)N_NODES * 4 * 4);
    float*    denom = (float*)carve((size_t)N_NODES * 4 * 4);
    // reuse of the (dead after GEMM1) XP region:
    float* e_exp = (float*)XP;                                     // EP*4 f32 = 26.4 MB
    float* c1    = (float*)((char*)XP + (size_t)32 * 1024 * 1024); // N*128 f32 = 25.6 MB
    _Float16* WgP[2] = {WgP0, WgP1};

    const int TB = 256;
    auto cdiv = [](long a, long b) { return (int)((a + b - 1) / b); };

    // pack weights to WMMA B fragments
    pack_b_f16<<<cdiv((long)IN_DIM * HID, TB), TB, 0, stream>>>(W_in,     WinP, IN_DIM, HID);
    pack_b_f16<<<cdiv((long)HID * HID, TB),   TB, 0, stream>>>(W_gat[0], WgP0, HID, HID);
    pack_b_f16<<<cdiv((long)HID * HID, TB),   TB, 0, stream>>>(W_gat[1], WgP1, HID, HID);
    pack_b_f16<<<cdiv((long)HID * C1DIM, TB), TB, 0, stream>>>(W_c1,     Wc1P, HID, C1DIM);

    // h = relu(x @ W_in + b_in)
    pack_a_f16<<<cdiv((long)N_NODES * IN_DIM, TB), TB, 0, stream>>>(x, XP, N_NODES, IN_DIM);
    {
        int mTiles = N_NODES / 16, nTiles = HID / 16, kChunks = IN_DIM / 32;
        int blocks = cdiv(mTiles, 8) * (nTiles / 4);
        gemm_wmma_f16<<<blocks, TB, 0, stream>>>(
            XP, WinP, b_in, h, mTiles, nTiles, kChunks, HID, 1);
    }

    // two GAT layers
    for (int l = 0; l < 2; ++l) {
        pack_a_f16<<<cdiv((long)N_NODES * HID, TB), TB, 0, stream>>>(h, HP, N_NODES, HID);
        {
            int mTiles = N_NODES / 16, nTiles = HID / 16, kChunks = HID / 32;
            int blocks = cdiv(mTiles, 8) * (nTiles / 4);
            gemm_wmma_f16<<<blocks, TB, 0, stream>>>(
                HP, WgP[l], nullptr, xh, mTiles, nTiles, kChunks, HID, 0);
        }
        att_dots<<<cdiv((long)N_NODES * 4 * 32, TB), TB, 0, stream>>>(
            xh, attS[l], attD[l], alsrc, aldst);

        fill_u32<<<cdiv((long)N_NODES * 4, TB), TB, 0, stream>>>(
            (uint32_t*)mmax, 0xFF800000u, (size_t)N_NODES * 4);          // -inf
        fill_u32<<<cdiv((long)N_NODES * 4, TB), TB, 0, stream>>>(
            (uint32_t*)denom, 0u, (size_t)N_NODES * 4);
        fill_u32<<<cdiv((long)N_NODES * HID, TB), TB, 0, stream>>>(
            (uint32_t*)outac, 0u, (size_t)N_NODES * HID);

        edge_softmax_max<<<cdiv((long)EP_EDGES, TB), TB, 0, stream>>>(
            ei, alsrc, aldst, mmax);
        edge_softmax_exp<<<cdiv((long)EP_EDGES, TB), TB, 0, stream>>>(
            ei, alsrc, aldst, mmax, e_exp, denom);
        edge_scatter<<<cdiv((long)EP_EDGES * 32, TB), TB, 0, stream>>>(
            ei, xh, e_exp, denom, outac);

        ln_relu_residual<<<cdiv((long)N_NODES * 32, TB), TB, 0, stream>>>(
            outac, b_gat[l], ln_g[l], ln_b[l], h);
    }

    // classifier: c1 = relu(h @ W_c1 + b_c1); logits = c1 @ W_c2 + b_c2
    pack_a_f16<<<cdiv((long)N_NODES * HID, TB), TB, 0, stream>>>(h, HP, N_NODES, HID);
    {
        int mTiles = N_NODES / 16, nTiles = C1DIM / 16, kChunks = HID / 32;
        int blocks = cdiv(mTiles, 8) * (nTiles / 4);
        gemm_wmma_f16<<<blocks, TB, 0, stream>>>(
            HP, Wc1P, b_c1, c1, mTiles, nTiles, kChunks, C1DIM, 1);
    }
    classifier2<<<cdiv((long)N_NODES, TB), TB, 0, stream>>>(c1, W_c2, b_c2, logits);
}
